// LatentEncoder_87643102642412
// MI455X (gfx1250) — compile-verified, matrix operand order checked by reading, not run on hardware
//
#include <hip/hip_runtime.h>
#include <hip/hip_bf16.h>

// ---------------------------------------------------------------------------
// Residual VQ encoder for MI455X (gfx1250, wave32, WMMA).
//
// x[32,4,64,64] f32 -> x_emb [N=131072, D=64]; 4 rounds of nearest-codebook
// (K=512) search; codes + loss + perplexity + used_codes.
//
// Design:
//  * scores = ||cb_k||^2 - 2 x.cb_k (||x||^2 constant per row -> dropped for
//    argmin, re-added for the loss).
//  * GEMM transposed: S^T = CB(16x64) x X^T(64x16) via 2 chained
//    v_wmma_f32_16x16x32_bf16, so argmin over K is lane-local over 8 C VGPRs
//    + one shfl_xor(16) merge of the two half-wave K partitions.
//  * Each wave carries RT=4 row-tiles of x_emb in registers (B operands), so
//    each staged codebook A-tile feeds 8 WMMAs and codebook L2 traffic drops
//    4x (256 blocks x 512KB = 128MB instead of 512MB).
//  * Codebook q is staged bf16 in LDS (72KB, 16B row pad for conflict-free
//    b128 operand loads); q+1's codebook is prefetched (global_prefetch_b8)
//    during q's scan.
// ---------------------------------------------------------------------------

typedef __attribute__((ext_vector_type(16))) __bf16 v16bf;
typedef __attribute__((ext_vector_type(8)))  float  v8f;

#define NTOT   131072
#define KC     512
#define DD     64
#define NQ     4
#define ROWB   144               // LDS bytes per codebook row: 128 data + 16 pad
#define RT     4                 // row-tiles (of 16 rows) per wave
#define RPB    (8 * RT * 16)     // rows per block = 512

union ABreg {
    v16bf        v;
    uint4        q[2];
    unsigned int u[8];
};

__device__ __forceinline__ unsigned short f2bf(float f) {
    unsigned int u = __float_as_uint(f);
    u += 0x7FFFu + ((u >> 16) & 1u);       // round-to-nearest-even
    return (unsigned short)(u >> 16);
}
__device__ __forceinline__ float bf2f(unsigned short h) {
    return __uint_as_float(((unsigned int)h) << 16);
}

// 16-bit WMMA operand layout (ISA 7.12.2): for a 16x32 operand, lane l holds
// row (l&15); with half = l>>4, vector elem e -> K index:
//   e in [0,8):  kk = 8*half + e
//   e in [8,16): kk = 16 + 8*half + (e-8)
// -> two contiguous 16B runs per 32-wide K chunk = two ds_load_b128.

__global__ __launch_bounds__(256)
void rvq_main_kernel(const float* __restrict__ x,
                     const float* __restrict__ Wm,
                     const float* __restrict__ bias,
                     const float* __restrict__ cbs,
                     float*        __restrict__ out_codes,  // NQ*NTOT floats
                     float*        __restrict__ ws_loss,    // [NQ]
                     unsigned int* __restrict__ ws_used)    // [NQ*16]
{
    __shared__ __align__(16) unsigned char lds_cb[KC * ROWB];   // 72 KB bf16
    __shared__ __align__(16) float         lds_cbn[KC];         // ||cb_k||^2
    __shared__ float                       lds_W[DD * 4];
    __shared__ float                       lds_b[DD];

    const int tid  = threadIdx.x;
    const int lane = tid & 31;
    const int wave = tid >> 5;
    const int half = lane >> 4;
    const int mrow = lane & 15;

    if (tid < DD * 4) lds_W[tid] = Wm[tid];
    if (tid < DD)     lds_b[tid] = bias[tid];
    __syncthreads();

    // ---- projection: build RT row-tiles of x_emb directly into B operands.
    ABreg B[RT][2];
    float xnorm[RT];
    int   nrow[RT];
    #pragma unroll
    for (int rt = 0; rt < RT; ++rt) {
        const int n = blockIdx.x * RPB + (wave * RT + rt) * 16 + mrow;
        nrow[rt] = n;
        float xv[4];
        const int bi = n >> 12;            // H*W = 4096
        const int hw = n & 4095;
        #pragma unroll
        for (int c = 0; c < 4; ++c) xv[c] = x[(bi * 4 + c) * 4096 + hw];

        float xn = 0.f;
        #pragma unroll
        for (int ch = 0; ch < 2; ++ch) {
            #pragma unroll
            for (int e2 = 0; e2 < 8; ++e2) {
                unsigned int pk = 0;
                #pragma unroll
                for (int p = 0; p < 2; ++p) {
                    const int e = 2 * e2 + p;
                    const int r = e >> 3, j = e & 7;
                    const int d = 32 * ch + 16 * r + 8 * half + j;
                    float v = lds_b[d];
                    #pragma unroll
                    for (int c = 0; c < 4; ++c) v += xv[c] * lds_W[d * 4 + c];
                    xn += v * v;
                    pk |= ((unsigned int)f2bf(v)) << (16 * p);
                }
                B[rt][ch].u[e2] = pk;
            }
        }
        // lane l holds half the d's; partner lane (l^16) has the complement.
        xnorm[rt] = xn + __shfl_xor(xn, 16);
    }

    for (int q = 0; q < NQ; ++q) {
        // ---- stage codebook q into LDS as bf16 (coalesced global reads)
        const float* cbq = cbs + q * KC * DD;
        for (int e = tid; e < KC * DD; e += 256) {
            const int k = e >> 6, d = e & 63;
            *(unsigned short*)(lds_cb + k * ROWB + d * 2) = f2bf(cbq[e]);
        }
        __syncthreads();
        for (int k = tid; k < KC; k += 256) {
            const unsigned short* row = (const unsigned short*)(lds_cb + k * ROWB);
            float s = 0.f;
            #pragma unroll 8
            for (int d = 0; d < DD; ++d) { const float v = bf2f(row[d]); s += v * v; }
            lds_cbn[k] = s;
        }
        __syncthreads();

        // ---- warm L2/L1 for the next quantizer's codebook while we compute
        if (q + 1 < NQ) {
            const char* nf = (const char*)(cbs + (q + 1) * KC * DD);
            #pragma unroll
            for (int i = 0; i < 4; ++i)
                __builtin_prefetch(nf + (tid + i * 256) * 128, 0, 1);
        }

        // ---- scan all 512 codes, 16 at a time; 8 WMMAs per staged A-tile
        float minv[RT];
        int   mini[RT];
        #pragma unroll
        for (int rt = 0; rt < RT; ++rt) { minv[rt] = 3.4e38f; mini[rt] = 0; }

        for (int t = 0; t < KC / 16; ++t) {
            const int kb = t * 16;
            const unsigned char* rb = lds_cb + (kb + mrow) * ROWB + 16 * half;
            ABreg A0, A1;
            A0.q[0] = *(const uint4*)(rb +  0);   // d =      8*half..+7
            A0.q[1] = *(const uint4*)(rb + 32);   // d = 16 + 8*half..+7
            A1.q[0] = *(const uint4*)(rb + 64);
            A1.q[1] = *(const uint4*)(rb + 96);

            const float* np = &lds_cbn[kb + 8 * half];
            const float4 n0 = *(const float4*)np;
            const float4 n1 = *(const float4*)(np + 4);
            const float nn[8] = {n0.x, n0.y, n0.z, n0.w, n1.x, n1.y, n1.z, n1.w};

            #pragma unroll
            for (int rt = 0; rt < RT; ++rt) {
                v8f c = {};
                c = __builtin_amdgcn_wmma_f32_16x16x32_bf16(false, A0.v, false,
                        B[rt][0].v, (short)0, c, false, false);
                c = __builtin_amdgcn_wmma_f32_16x16x32_bf16(false, A1.v, false,
                        B[rt][1].v, (short)0, c, false, false);
                #pragma unroll
                for (int j = 0; j < 8; ++j) {
                    const float s = nn[j] - 2.0f * c[j];
                    if (s < minv[rt]) { minv[rt] = s; mini[rt] = kb + 8 * half + j; }
                }
            }
        }

        // ---- per-tile finalize: merge halves, emit codes, update x_emb
        float lacc = 0.f;
        #pragma unroll
        for (int rt = 0; rt < RT; ++rt) {
            const float ov = __shfl_xor(minv[rt], 16);
            const int   oi = __shfl_xor(mini[rt], 16);
            if (ov < minv[rt] || (ov == minv[rt] && oi < mini[rt])) {
                minv[rt] = ov; mini[rt] = oi;
            }
            if (lane < 16) {
                out_codes[q * NTOT + nrow[rt]] = (float)mini[rt];
                atomicOr(&ws_used[q * 16 + (mini[rt] >> 5)], 1u << (mini[rt] & 31));
            }
            // ||x - cb||^2 = ||x||^2 + min score
            lacc += fmaxf(xnorm[rt] + minv[rt], 0.f);

            if (q + 1 < NQ) {   // x_emb <- cb[code], straight from staged LDS
                const unsigned char* rb2 = lds_cb + mini[rt] * ROWB + 16 * half;
                B[rt][0].q[0] = *(const uint4*)(rb2 +  0);
                B[rt][0].q[1] = *(const uint4*)(rb2 + 32);
                B[rt][1].q[0] = *(const uint4*)(rb2 + 64);
                B[rt][1].q[1] = *(const uint4*)(rb2 + 96);
                xnorm[rt] = lds_cbn[mini[rt]];
            }
        }
        // every row is represented in 2 lanes -> halve after wave reduction
        #pragma unroll
        for (int s = 1; s < 32; s <<= 1) lacc += __shfl_xor(lacc, s);
        if (lane == 0) atomicAdd(&ws_loss[q], lacc * 0.5f);

        __syncthreads();   // codebook LDS is overwritten next iteration
    }
}

__global__ void rvq_init_kernel(float* ws_loss, unsigned int* ws_used) {
    const int t = threadIdx.x;
    if (t < NQ)      ws_loss[t] = 0.f;
    if (t < NQ * 16) ws_used[t] = 0u;
}

__global__ void rvq_finalize_kernel(const float* __restrict__ ws_loss,
                                    const unsigned int* __restrict__ ws_used,
                                    float* __restrict__ out) {
    if (threadIdx.x == 0) {
        float lsum = 0.f;
        for (int q = 0; q < NQ; ++q) lsum += ws_loss[q];
        // per-q loss = 1.25 * sum_d2 / (N*D); final = mean over q
        const float loss = lsum * 1.25f / ((float)NQ * (float)NTOT * (float)DD);
        float perp = 0.f;
        int uni = 0;
        for (int w = 0; w < 16; ++w) {
            unsigned int u = 0;
            for (int q = 0; q < NQ; ++q) {
                const unsigned int v = ws_used[q * 16 + w];
                perp += (float)__popc(v);
                u |= v;
            }
            uni += __popc(u);
        }
        out[NQ * NTOT + 0] = loss;
        out[NQ * NTOT + 1] = perp * (1.0f / (float)NQ);
        out[NQ * NTOT + 2] = (float)uni;
    }
}

extern "C" void kernel_launch(void* const* d_in, const int* in_sizes, int n_in,
                              void* d_out, int out_size, void* d_ws, size_t ws_size,
                              hipStream_t stream) {
    const float* x    = (const float*)d_in[0];   // [32,4,64,64]
    const float* Wm   = (const float*)d_in[1];   // [64,4]
    const float* bias = (const float*)d_in[2];   // [64]
    const float* cbs  = (const float*)d_in[3];   // [4,512,64]
    float* out = (float*)d_out;                  // NQ*NTOT codes + loss + perp + used

    float*        ws_loss = (float*)d_ws;
    unsigned int* ws_used = (unsigned int*)((char*)d_ws + 64);

    rvq_init_kernel<<<1, 64, 0, stream>>>(ws_loss, ws_used);
    rvq_main_kernel<<<NTOT / RPB, 256, 0, stream>>>(x, Wm, bias, cbs, out,
                                                    ws_loss, ws_used);
    rvq_finalize_kernel<<<1, 32, 0, stream>>>(ws_loss, ws_used, out);
}
